// MoM_8383776161860
// MI455X (gfx1250) — matrix operand order repaired and checked by reading, not compiled, not run on hardware
//
#include <hip/hip_runtime.h>

// ---------------------------------------------------------------------------
// MoM forward for MI455X (gfx1250, wave32, WMMA).
// Phase 1: one big bf16 WMMA GEMM for all projections (hoisted out of scan):
//          P(8192 x 1280) = X(8192 x 1024) @ Wcat^T + bias
//          cols [0,576)=k, [576,1152)=v, [1152,1216)=q, [1216,1224)=gate
// Phase 2: sequential scan, 1 workgroup per batch, M state in LDS (144KB).
// ---------------------------------------------------------------------------

typedef __bf16 bf16;
typedef __attribute__((ext_vector_type(8)))  bf16  v8bf;
typedef __attribute__((ext_vector_type(16))) bf16  v16bf;
typedef __attribute__((ext_vector_type(8)))  float v8f;

#define SEQ    256
#define BATCH  32
#define IN_DIM 1024
#define HID    64
#define NMEM   8
#define NM1    9
#define NPAD   1280          // 1224 projection cols padded to 80*16
#define KCOL   0
#define VCOL   576
#define QCOL   1152
#define GCOL   1216
#define MROWS  (SEQ*BATCH)   // 8192
#define MSLOT  (HID*HID)     // 4096

// ---------------- prep: fp32 -> bf16 copy of X ----------------
__global__ void cvt_x_kernel(const float* __restrict__ X, bf16* __restrict__ Xb, int n) {
    int i = blockIdx.x * blockDim.x + threadIdx.x;
    if (i < n) Xb[i] = (bf16)X[i];
}

// ---------------- prep: pack [Wk;Wv;Wq;Wg] -> bf16 Wcat (1280x1024) + bias ----
__global__ void pack_w_kernel(const float* __restrict__ Wk, const float* __restrict__ Wv,
                              const float* __restrict__ Wq, const float* __restrict__ Wg,
                              const float* __restrict__ bk, const float* __restrict__ bv,
                              const float* __restrict__ bq, const float* __restrict__ bg,
                              bf16* __restrict__ Wb, float* __restrict__ bias) {
    int i = blockIdx.x * blockDim.x + threadIdx.x;   // over NPAD*IN_DIM
    int row = i >> 10;
    int col = i & 1023;
    float w = 0.0f;
    if      (row < 576)  w = Wk[row * 1024 + col];
    else if (row < 1152) w = Wv[(row - 576) * 1024 + col];
    else if (row < 1216) w = Wq[(row - 1152) * 1024 + col];
    else if (row < 1224) w = Wg[(row - 1216) * 1024 + col];
    Wb[i] = (bf16)w;
    if (i < NPAD) {
        float b = 0.0f;
        if      (i < 576)  b = bk[i];
        else if (i < 1152) b = bv[i - 576];
        else if (i < 1216) b = bq[i - 1152];
        else if (i < 1224) b = bg[i - 1216];
        bias[i] = b;
    }
}

// ---------------- GEMM: P = Xb @ Wb^T + bias, bf16 WMMA, f32 accum ----------
// One wave computes a 16(M) x 64(N) strip: 4 accumulators, A regs reused x4.
__global__ __launch_bounds__(256)
void gemm_bf16_kernel(const bf16* __restrict__ A,    // 8192 x 1024 row-major
                      const bf16* __restrict__ B,    // 1280 x 1024 row-major (row = out col)
                      const float* __restrict__ bias,
                      float* __restrict__ P) {       // 8192 x 1280
    const int lane = threadIdx.x & 31;
    const int wave = threadIdx.x >> 5;
    const int w    = blockIdx.x * 8 + wave;          // 10240 waves total
    const int mt   = w / 20;                         // 0..511  (row tile)
    const int nq   = w % 20;                         // 0..19   (4 col tiles each)
    const int half = lane >> 4;                      // 0|1 (lane group)
    const int l15  = lane & 15;

    // A operand, contiguous k order chosen here (same as B): lanes 0-15 hold K0..15; lanes 16-31 K16..31
    const int row = mt * 16 + l15;
    const bf16* arow = A + (size_t)row * IN_DIM + (half ? 16 : 0);
    // B operand, same contiguous k order as A: lanes 0-15 hold K0..15 of col, lanes 16-31 K16..31
    const int kb_off = half ? 16 : 0;
    const int c0 = (nq * 4 + 0) * 16 + l15;
    const int c1 = (nq * 4 + 1) * 16 + l15;
    const int c2 = (nq * 4 + 2) * 16 + l15;
    const int c3 = (nq * 4 + 3) * 16 + l15;
    const bf16* b0 = B + (size_t)c0 * IN_DIM + kb_off;
    const bf16* b1 = B + (size_t)c1 * IN_DIM + kb_off;
    const bf16* b2 = B + (size_t)c2 * IN_DIM + kb_off;
    const bf16* b3 = B + (size_t)c3 * IN_DIM + kb_off;

    v8f acc0 = {}, acc1 = {}, acc2 = {}, acc3 = {};

    #pragma unroll 4
    for (int k0 = 0; k0 < IN_DIM; k0 += 32) {
        v8bf alo = *(const v8bf*)(arow + k0);
        v8bf ahi = *(const v8bf*)(arow + k0 + 8);
        v16bf a = __builtin_shufflevector(alo, ahi,
                      0, 1, 2, 3, 4, 5, 6, 7, 8, 9, 10, 11, 12, 13, 14, 15);
        v16bf vb0 = *(const v16bf*)(b0 + k0);
        v16bf vb1 = *(const v16bf*)(b1 + k0);
        v16bf vb2 = *(const v16bf*)(b2 + k0);
        v16bf vb3 = *(const v16bf*)(b3 + k0);
        acc0 = __builtin_amdgcn_wmma_f32_16x16x32_bf16(false, a, false, vb0, (short)0, acc0, false, false);
        acc1 = __builtin_amdgcn_wmma_f32_16x16x32_bf16(false, a, false, vb1, (short)0, acc1, false, false);
        acc2 = __builtin_amdgcn_wmma_f32_16x16x32_bf16(false, a, false, vb2, (short)0, acc2, false, false);
        acc3 = __builtin_amdgcn_wmma_f32_16x16x32_bf16(false, a, false, vb3, (short)0, acc3, false, false);
    }

    // C/D layout: VGPR v holds (M = v + 8*half, N = lane&15)
    const int rbase = mt * 16 + half * 8;
    const float bb0 = bias[c0], bb1 = bias[c1], bb2 = bias[c2], bb3 = bias[c3];
    #pragma unroll
    for (int v = 0; v < 8; ++v) {
        const size_t r = (size_t)(rbase + v) * NPAD;
        P[r + c0] = acc0[v] + bb0;
        P[r + c1] = acc1[v] + bb1;
        P[r + c2] = acc2[v] + bb2;
        P[r + c3] = acc3[v] + bb3;
    }
}

// ---------------- scan: 1 block per batch, M state (9x64x64 f32) in LDS -----
__global__ __launch_bounds__(256)
void scan_kernel(const float* __restrict__ P,   // 8192 x 1280
                 const float* __restrict__ M0,  // B x 9 x 64 x 64
                 float* __restrict__ out) {     // [SEQ*B*64 outputs | B*9*64*64 M_final]
    extern __shared__ float Msh[];              // NM1*MSLOT floats = 144KB
    __shared__ float ksel[3][HID];
    __shared__ float vsel[3][HID];
    __shared__ float qv[HID];
    __shared__ float red[4][HID];
    __shared__ float gsh[2];
    __shared__ int   idxsh[3];

    const int b   = blockIdx.x;
    const int tid = threadIdx.x;

    for (int e = tid; e < NM1 * MSLOT; e += 256)
        Msh[e] = M0[(size_t)b * NM1 * MSLOT + e];
    __syncthreads();

    for (int t = 0; t < SEQ; ++t) {
        const int r = t * BATCH + b;
        const float* prow = P + (size_t)r * NPAD;

        // gating: top-2 of logits (softmax->topk->renorm == 2-way softmax of the 2 logits)
        if (tid == 0) {
            float l[NMEM];
            #pragma unroll
            for (int m = 0; m < NMEM; ++m) l[m] = prow[GCOL + m];
            int m0 = 0;
            #pragma unroll
            for (int m = 1; m < NMEM; ++m) if (l[m] > l[m0]) m0 = m;
            int m1 = (m0 == 0) ? 1 : 0;
            #pragma unroll
            for (int m = 0; m < NMEM; ++m) if (m != m0 && l[m] > l[m1]) m1 = m;
            float e1 = __expf(l[m1] - l[m0]);
            float g0 = 1.0f / (1.0f + e1);
            idxsh[0] = 0; idxsh[1] = m0; idxsh[2] = m1;
            gsh[0] = g0;  gsh[1] = 1.0f - g0;
        }
        __syncthreads();

        // gather routed k/v rows and q
        if (tid < 192) {
            const int s = tid >> 6, h = tid & 63;
            ksel[s][h] = prow[KCOL + idxsh[s] * HID + h];
            vsel[s][h] = prow[VCOL + idxsh[s] * HID + h];
        }
        if (tid < HID) qv[tid] = prow[QCOL + tid];
        __syncthreads();

        // scatter-add 3 outer products; each thread owns a fixed 16-element slice,
        // so duplicate slot indices accumulate correctly (sequential in-thread).
        {
            const int e0 = tid * 16;
            #pragma unroll
            for (int s = 0; s < 3; ++s) {
                float* Mslot = Msh + idxsh[s] * MSLOT;
                #pragma unroll
                for (int u = 0; u < 16; ++u) {
                    const int e = e0 + u;
                    Mslot[e] += ksel[s][e >> 6] * vsel[s][e & 63];
                }
            }
        }
        __syncthreads();

        // o_j = sum_i q_i * (M[0] + g0*M[m0] + g1*M[m1])[i][j], split over 4 partials
        {
            const int j = tid & 63, part = tid >> 6;
            const float g0 = gsh[0], g1 = gsh[1];
            const float* Ma = Msh + idxsh[1] * MSLOT;
            const float* Mb = Msh + idxsh[2] * MSLOT;
            float acc = 0.0f;
            #pragma unroll
            for (int u = 0; u < 16; ++u) {
                const int i = part * 16 + u;
                const float comb = Msh[i * HID + j] + g0 * Ma[i * HID + j] + g1 * Mb[i * HID + j];
                acc = __builtin_fmaf(qv[i], comb, acc);
            }
            red[part][j] = acc;
        }
        __syncthreads();
        if (tid < HID)
            out[(size_t)r * HID + tid] = red[0][tid] + red[1][tid] + red[2][tid] + red[3][tid];
        __syncthreads();
    }

    // M_final
    for (int e = tid; e < NM1 * MSLOT; e += 256)
        out[(size_t)SEQ * BATCH * HID + (size_t)b * NM1 * MSLOT + e] = Msh[e];
}

// ---------------------------------------------------------------------------
extern "C" void kernel_launch(void* const* d_in, const int* in_sizes, int n_in,
                              void* d_out, int out_size, void* d_ws, size_t ws_size,
                              hipStream_t stream) {
    const float* X  = (const float*)d_in[0];
    const float* M0 = (const float*)d_in[1];
    const float* Wk = (const float*)d_in[2];
    const float* bk = (const float*)d_in[3];
    const float* Wv = (const float*)d_in[4];
    const float* bv = (const float*)d_in[5];
    const float* Wg = (const float*)d_in[6];
    const float* bg = (const float*)d_in[7];
    const float* Wq = (const float*)d_in[8];
    const float* bq = (const float*)d_in[9];
    float* out = (float*)d_out;

    // workspace layout (bytes):
    //   [0,           16777216) Xbf   8192*1024 bf16
    //   [16777216,    19398656) Wbf   1280*1024 bf16
    //   [19398656,    19403776) bias  1280 f32
    //   [19403776,    61346816) P     8192*1280 f32
    char* ws = (char*)d_ws;
    bf16*  Xb   = (bf16*)(ws);
    bf16*  Wb   = (bf16*)(ws + 16777216);
    float* bias = (float*)(ws + 19398656);
    float* P    = (float*)(ws + 19403776);

    cvt_x_kernel<<<(MROWS * IN_DIM) / 256, 256, 0, stream>>>(X, Xb, MROWS * IN_DIM);
    pack_w_kernel<<<(NPAD * IN_DIM) / 256, 256, 0, stream>>>(Wk, Wv, Wq, Wg, bk, bv, bq, bg, Wb, bias);
    gemm_bf16_kernel<<<1280, 256, 0, stream>>>(Xb, Wb, bias, P);   // 10240 waves, 16x64 strip each
    scan_kernel<<<BATCH, 256, NM1 * MSLOT * (int)sizeof(float), stream>>>(P, M0, out);
}
